// MultiHeadAttention_14370960573094
// MI455X (gfx1250) — matrix operand order repaired and checked by reading, not compiled, not run on hardware
//
#include <hip/hip_runtime.h>
#include <hip/hip_bf16.h>

typedef __attribute__((ext_vector_type(16))) __bf16 v16bf;
typedef __attribute__((ext_vector_type(8)))  __bf16 bf16x8;
typedef __attribute__((ext_vector_type(4)))  __bf16 bf16x4;
typedef __attribute__((ext_vector_type(8)))  float  v8f;
typedef __attribute__((ext_vector_type(4)))  float  f32x4;

constexpr int G_REAL   = 131;    // graph size
constexpr int GP       = 160;    // padded rows per batch (multiple of 32)
constexpr int SS       = 144;    // S row/col stride (scores padded to 9x9 tiles)
constexpr int BATCH    = 512;
constexpr int DIM      = 512;
constexpr int INTERNAL = 80;
constexpr int MPAD     = BATCH * GP;         // 81920 padded rows

static __device__ __forceinline__ v8f wmma_bf16(v16bf a, v16bf b, v8f c) {
  return __builtin_amdgcn_wmma_f32_16x16x32_bf16(false, a, false, b, (short)0, c,
                                                 false, false);
}

// A fragment (16 rows x 32 K), bf16 row-major, leading dim ld.
// ISA layout: lane=M%16; lanes0-15 elems0-7=K0..7, elems8-15=K16..23;
// lanes16-31 elems0-7=K8..15, elems8-15=K24..31.
static __device__ __forceinline__ v16bf load_a(const __bf16* base, int ld,
                                               int m0, int k0) {
  int lane = threadIdx.x & 31, lm = lane & 15, lh = lane >> 4;
  const bf16x8* p = (const bf16x8*)(base + (size_t)(m0 + lm) * ld + k0 + lh * 8);
  bf16x8 lo = p[0];
  bf16x8 hi = p[2];  // +16 elements
  v16bf r;
#pragma unroll
  for (int e = 0; e < 8; ++e) { r[e] = lo[e]; r[e + 8] = hi[e]; }
  return r;
}

// B fragment (32 K x 16 N) from row-major (N,K) matrix -> computes A*M^T.
// Lane layout: lane=N%16; lanes0-15 elems0-7=K0..7, elems8-15=K16..23; lanes16-31 elems0-7=K8..15, elems8-15=K24..31.
static __device__ __forceinline__ v16bf load_b(const __bf16* base, int ld,
                                               int n0, int k0) {
  int lane = threadIdx.x & 31, lm = lane & 15, lh = lane >> 4;
  const bf16x8* p = (const bf16x8*)(base + (size_t)(n0 + lm) * ld + k0 + lh * 8);
  bf16x8 lo = p[0], hi = p[2];
  v16bf r;
#pragma unroll
  for (int e = 0; e < 8; ++e) { r[e] = lo[e]; r[e + 8] = hi[e]; }
  return r;
}

static __device__ __forceinline__ void mm_step(v8f (&acc)[2][4], const v16bf (&a)[2],
                                               const v16bf (&b)[4]) {
#pragma unroll
  for (int t = 0; t < 2; ++t)
#pragma unroll
    for (int j = 0; j < 4; ++j) acc[t][j] = wmma_bf16(a[t], b[j], acc[t][j]);
}

// ---------- prep: convert fp32 -> bf16 with padded-row layout (zeros for g>=131) ------
__global__ void k_cvt_pad(const float* __restrict__ src, __bf16* __restrict__ dst) {
  int r = blockIdx.x;              // padded row 0..MPAD-1
  int c = threadIdx.x << 2;        // 128 threads * 4 cols
  int b = r / GP, g = r - b * GP;
  bf16x4 o;
  if (g < G_REAL) {
    f32x4 v = *(const f32x4*)(src + (size_t)(b * G_REAL + g) * DIM + c);
#pragma unroll
    for (int e = 0; e < 4; ++e) o[e] = (__bf16)v[e];
  } else {
#pragma unroll
    for (int e = 0; e < 4; ++e) o[e] = (__bf16)0.0f;
  }
  *(bf16x4*)(dst + (size_t)r * DIM + c) = o;
}

// ---------- prep: plain fp32 -> bf16 (weights, 512x512) ----------
__global__ void k_cvt(const float* __restrict__ src, __bf16* __restrict__ dst) {
  int idx = (blockIdx.x * 256 + threadIdx.x) << 2;
  f32x4 v = *(const f32x4*)(src + idx);
  bf16x4 o;
#pragma unroll
  for (int e = 0; e < 4; ++e) o[e] = (__bf16)v[e];
  *(bf16x4*)(dst + idx) = o;
}

// ---------- generic GEMM: (MPAD x 512) x (512 x 512)^T, 8 waves, wave tile 32x64 ------
// MODE 0: store bf16 rows (C: MPAD x 512)
// MODE 1: store transposed Vt[b][n][g] (vector 16B store per lane)
// MODE 2: f32 out + bias, un-pad rows (out: (B*131) x 512)
template <int MODE>
__global__ void k_gemm(const __bf16* __restrict__ A, const __bf16* __restrict__ Bw,
                       const float* __restrict__ bias, void* __restrict__ Cout) {
  int wid = threadIdx.x >> 5;
  int lane = threadIdx.x & 31, lm = lane & 15, lh = lane >> 4;
  int wm = wid & 3, wn = wid >> 2;
  int m0w = blockIdx.x * 128 + wm * 32;
  int n0w = blockIdx.y * 128 + wn * 64;

  v8f acc[2][4];
#pragma unroll
  for (int t = 0; t < 2; ++t)
#pragma unroll
    for (int j = 0; j < 4; ++j) acc[t][j] = {};

  v16bf ac[2], bc[4];
#pragma unroll
  for (int t = 0; t < 2; ++t) ac[t] = load_a(A, DIM, m0w + t * 16, 0);
#pragma unroll
  for (int j = 0; j < 4; ++j) bc[j] = load_b(Bw, DIM, n0w + j * 16, 0);

  for (int k0 = 32; k0 < DIM; k0 += 32) {
    v16bf an[2], bn[4];
#pragma unroll
    for (int t = 0; t < 2; ++t) an[t] = load_a(A, DIM, m0w + t * 16, k0);
#pragma unroll
    for (int j = 0; j < 4; ++j) bn[j] = load_b(Bw, DIM, n0w + j * 16, k0);
    mm_step(acc, ac, bc);
#pragma unroll
    for (int t = 0; t < 2; ++t) ac[t] = an[t];
#pragma unroll
    for (int j = 0; j < 4; ++j) bc[j] = bn[j];
  }
  mm_step(acc, ac, bc);

#pragma unroll
  for (int t = 0; t < 2; ++t) {
#pragma unroll
    for (int j = 0; j < 4; ++j) {
      int tm = m0w + t * 16, tn = n0w + j * 16;
      if (MODE == 0) {
        __bf16* C = (__bf16*)Cout;
#pragma unroll
        for (int i = 0; i < 8; ++i)
          C[(size_t)(tm + lh * 8 + i) * DIM + tn + lm] = (__bf16)acc[t][j][i];
      } else if (MODE == 1) {
        __bf16* C = (__bf16*)Cout;
        int bb = tm / GP, g0 = tm - bb * GP;  // 32-row wave tile stays in one batch
        bf16x8 v;
#pragma unroll
        for (int i = 0; i < 8; ++i) v[i] = (__bf16)acc[t][j][i];
        *(bf16x8*)(C + ((size_t)bb * DIM + tn + lm) * GP + g0 + lh * 8) = v;
      } else {
        float* out = (float*)Cout;
        float bv = bias[tn + lm];
#pragma unroll
        for (int i = 0; i < 8; ++i) {
          int rp = tm + lh * 8 + i;
          int bb = rp / GP, g = rp - bb * GP;
          if (g < G_REAL)
            out[(size_t)(bb * G_REAL + g) * DIM + tn + lm] = acc[t][j][i] + bv;
        }
      }
    }
  }
}

// ---------- scores: S(b,i,j) = norm * sum_k Q[b,i,k] K[b,j,k]; 3 tj tiles per wave ----
__global__ void k_scores(const __bf16* __restrict__ Q, const __bf16* __restrict__ K,
                         float* __restrict__ S) {
  int b = blockIdx.x;
  int wid = threadIdx.x >> 5, lane = threadIdx.x & 31, lm = lane & 15, lh = lane >> 4;
  const __bf16* Qb = Q + (size_t)b * GP * DIM;
  const __bf16* Kb = K + (size_t)b * GP * DIM;
  float* Sb = S + (size_t)b * SS * SS;
  const float norm = 0.04419417382415922f;  // 1/sqrt(512)
  for (int job = wid; job < 27; job += 8) {  // 9 ti x 3 tj-groups
    int ti = job / 3, tj0 = (job - ti * 3) * 3;
    v8f acc[3];
#pragma unroll
    for (int j = 0; j < 3; ++j) acc[j] = {};
    v16bf ac = load_a(Qb, DIM, ti * 16, 0);
    v16bf bc[3];
#pragma unroll
    for (int j = 0; j < 3; ++j) bc[j] = load_b(Kb, DIM, (tj0 + j) * 16, 0);
    for (int k0 = 32; k0 < DIM; k0 += 32) {
      v16bf an = load_a(Qb, DIM, ti * 16, k0);
      v16bf bn[3];
#pragma unroll
      for (int j = 0; j < 3; ++j) bn[j] = load_b(Kb, DIM, (tj0 + j) * 16, k0);
#pragma unroll
      for (int j = 0; j < 3; ++j) acc[j] = wmma_bf16(ac, bc[j], acc[j]);
      ac = an;
#pragma unroll
      for (int j = 0; j < 3; ++j) bc[j] = bn[j];
    }
#pragma unroll
    for (int j = 0; j < 3; ++j) acc[j] = wmma_bf16(ac, bc[j], acc[j]);
#pragma unroll
    for (int j = 0; j < 3; ++j)
#pragma unroll
      for (int i = 0; i < 8; ++i)
        Sb[(size_t)(ti * 16 + lh * 8 + i) * SS + (tj0 + j) * 16 + lm] =
            acc[j][i] * norm;
  }
}

// ---------- masked softmax (reference semantics) -> P (bf16, GP x GP per batch) ------
__global__ void k_softmax(const float* __restrict__ S, const int* __restrict__ mask,
                          const int* __restrict__ evaluate, __bf16* __restrict__ P) {
  int wid = threadIdx.x >> 5, lane = threadIdx.x & 31;
  int rid = blockIdx.x * 8 + wid;        // row id in (BATCH*GP)
  int b = rid / GP, i = rid - b * GP;
  __bf16* Pr = P + (size_t)rid * GP;
  if (i >= G_REAL) {
#pragma unroll
    for (int k = 0; k < 5; ++k) Pr[lane + 32 * k] = (__bf16)0.0f;
    return;
  }
  const float* Sr = S + ((size_t)b * SS + i) * SS;
  const int* mb = mask + b * G_REAL;
  const float NEG_INF = -__builtin_inff();
  float fill = (evaluate[0] != 0) ? NEG_INF : -30.0f;
  bool mid_i = (i >= INTERNAL) && (i < G_REAL - 1);

  float s[5]; bool mk[5];
#pragma unroll
  for (int k = 0; k < 5; ++k) {
    int j = lane + 32 * k;               // 0..159
    if (j < G_REAL) {
      bool m = (mb[j] > 0) || (mid_i && (j >= INTERNAL) && (j < G_REAL - 1));
      mk[k] = m;
      s[k] = m ? fill : Sr[j];
    } else { mk[k] = true; s[k] = NEG_INF; }
  }
  float mx = s[0];
#pragma unroll
  for (int k = 1; k < 5; ++k) mx = fmaxf(mx, s[k]);
#pragma unroll
  for (int d = 16; d >= 1; d >>= 1) mx = fmaxf(mx, __shfl_xor(mx, d, 32));
  float e[5]; float sum = 0.0f;
#pragma unroll
  for (int k = 0; k < 5; ++k) { e[k] = __expf(s[k] - mx); sum += e[k]; }
#pragma unroll
  for (int d = 16; d >= 1; d >>= 1) sum += __shfl_xor(sum, d, 32);
  float inv = 1.0f / sum;
#pragma unroll
  for (int k = 0; k < 5; ++k) {
    int j = lane + 32 * k;
    Pr[j] = (__bf16)(mk[k] ? 0.0f : e[k] * inv);  // zero masked + padded (attnc)
  }
}

// ---------- heads = P * V (Vt[b][n][g]); same 128x128 block shape, K=160 ------------
__global__ void k_pv(const __bf16* __restrict__ P, const __bf16* __restrict__ Vt,
                     __bf16* __restrict__ H) {
  int wid = threadIdx.x >> 5;
  int lane = threadIdx.x & 31, lm = lane & 15, lh = lane >> 4;
  int wm = wid & 3, wn = wid >> 2;
  int m0w = blockIdx.x * 128 + wm * 32;
  int n0w = blockIdx.y * 128 + wn * 64;
  int bb = m0w / GP;                                  // wave tile within one batch
  const __bf16* Vb = Vt + (size_t)bb * DIM * GP;      // (512, 160)

  v8f acc[2][4];
#pragma unroll
  for (int t = 0; t < 2; ++t)
#pragma unroll
    for (int j = 0; j < 4; ++j) acc[t][j] = {};

  v16bf ac[2], bc[4];
#pragma unroll
  for (int t = 0; t < 2; ++t) ac[t] = load_a(P, GP, m0w + t * 16, 0);
#pragma unroll
  for (int j = 0; j < 4; ++j) bc[j] = load_b(Vb, GP, n0w + j * 16, 0);

  for (int k0 = 32; k0 < GP; k0 += 32) {
    v16bf an[2], bn[4];
#pragma unroll
    for (int t = 0; t < 2; ++t) an[t] = load_a(P, GP, m0w + t * 16, k0);
#pragma unroll
    for (int j = 0; j < 4; ++j) bn[j] = load_b(Vb, GP, n0w + j * 16, k0);
    mm_step(acc, ac, bc);
#pragma unroll
    for (int t = 0; t < 2; ++t) ac[t] = an[t];
#pragma unroll
    for (int j = 0; j < 4; ++j) bc[j] = bn[j];
  }
  mm_step(acc, ac, bc);

#pragma unroll
  for (int t = 0; t < 2; ++t)
#pragma unroll
    for (int j = 0; j < 4; ++j)
#pragma unroll
      for (int i = 0; i < 8; ++i)
        H[(size_t)(m0w + t * 16 + lh * 8 + i) * DIM + n0w + j * 16 + lm] =
            (__bf16)acc[t][j][i];
}

extern "C" void kernel_launch(void* const* d_in, const int* in_sizes, int n_in,
                              void* d_out, int out_size, void* d_ws, size_t ws_size,
                              hipStream_t stream) {
  (void)in_sizes; (void)n_in; (void)out_size; (void)ws_size;
  const float* q        = (const float*)d_in[0];
  const int*   mask     = (const int*)d_in[1];
  const int*   evaluate = (const int*)d_in[3];
  const float* h        = (const float*)d_in[4];
  const float* Wq       = (const float*)d_in[5];
  const float* Wk       = (const float*)d_in[6];
  const float* Wv       = (const float*)d_in[7];
  const float* Wo       = (const float*)d_in[8];
  const float* bo       = (const float*)d_in[9];
  float* out = (float*)d_out;

  char* ws = (char*)d_ws;
  const size_t SZ_BIG = (size_t)MPAD * DIM * sizeof(__bf16);       // 83,886,080
  const size_t SZ_W   = (size_t)DIM * DIM * sizeof(__bf16);        // 524,288
  const size_t SZ_S   = (size_t)BATCH * SS * SS * sizeof(float);   // 42,467,328

  // Region reuse across kernel-boundary lifetimes:
  //  R0: q_bf  -> heads (q_bf dead after Q projection)
  //  R1: h_bf  -> S + P (h_bf dead after V projection)
  __bf16* qbf = (__bf16*)(ws);                       // R0
  __bf16* hbf = (__bf16*)(ws + SZ_BIG);              // R1
  __bf16* Qb  = (__bf16*)(ws + 2 * SZ_BIG);
  __bf16* Kb  = (__bf16*)(ws + 3 * SZ_BIG);
  __bf16* Vt  = (__bf16*)(ws + 4 * SZ_BIG);
  __bf16* Wqb = (__bf16*)(ws + 5 * SZ_BIG);
  __bf16* Wkb = (__bf16*)(ws + 5 * SZ_BIG + SZ_W);
  __bf16* Wvb = (__bf16*)(ws + 5 * SZ_BIG + 2 * SZ_W);
  __bf16* Wob = (__bf16*)(ws + 5 * SZ_BIG + 3 * SZ_W);
  float*  S   = (float*)hbf;                         // aliases R1
  __bf16* P   = (__bf16*)(ws + SZ_BIG + SZ_S);       // R1 + 42.5MB
  __bf16* H   = (__bf16*)qbf;                        // aliases R0

  dim3 blk(256);
  // prep: fp32 -> bf16 (+row padding for activations)
  k_cvt_pad<<<dim3(MPAD), dim3(128), 0, stream>>>(q, qbf);
  k_cvt_pad<<<dim3(MPAD), dim3(128), 0, stream>>>(h, hbf);
  k_cvt<<<dim3(256), blk, 0, stream>>>(Wq, Wqb);
  k_cvt<<<dim3(256), blk, 0, stream>>>(Wk, Wkb);
  k_cvt<<<dim3(256), blk, 0, stream>>>(Wv, Wvb);
  k_cvt<<<dim3(256), blk, 0, stream>>>(Wo, Wob);
  // projections (MPAD/128 = 640 M-blocks, 512/128 = 4 N-blocks)
  k_gemm<0><<<dim3(640, 4), blk, 0, stream>>>(qbf, Wqb, nullptr, Qb);
  k_gemm<0><<<dim3(640, 4), blk, 0, stream>>>(hbf, Wkb, nullptr, Kb);
  k_gemm<1><<<dim3(640, 4), blk, 0, stream>>>(hbf, Wvb, nullptr, Vt);
  // attention
  k_scores<<<dim3(BATCH), blk, 0, stream>>>(Qb, Kb, S);
  k_softmax<<<dim3(MPAD / 8), blk, 0, stream>>>(S, mask, evaluate, P);
  k_pv<<<dim3(640, 4), blk, 0, stream>>>(P, Vt, H);
  // output projection + bias, un-padded f32 store
  k_gemm<2><<<dim3(640, 4), blk, 0, stream>>>(H, Wob, bo, out);
}
